// GAT_2layer_76836964925934
// MI455X (gfx1250) — matrix operand order repaired and checked by reading, not compiled, not run on hardware
//
#include <hip/hip_runtime.h>
#include <cstdint>
#include <cstddef>

// ---------------------------------------------------------------------------
// GATv2 x2 on MI455X (gfx1250). Memory-bound (~3 GB HBM traffic vs ~62 GFLOP
// of GEMM). All GEMMs run on v_wmma_f32_16x16x32_bf16 (wave32, f32 accum).
// GEMM operands are pre-converted to bf16 once (they are re-read many times),
// tiles are double-buffered through LDS so next-tile global b128 loads overlap
// the current tile's WMMA burst. Edge-space e1 ([Etot,2048] f32) is chunked
// (8192 edges) so it never materializes fully.
// ---------------------------------------------------------------------------

typedef __attribute__((ext_vector_type(16))) __bf16 bf16x16;
typedef __attribute__((ext_vector_type(8)))  __bf16 bf16x8;
typedef __attribute__((ext_vector_type(4)))  __bf16 bf16x4;
typedef __attribute__((ext_vector_type(8)))  float  f32x8;
typedef __attribute__((ext_vector_type(4)))  float  f32x4;

#define NNODES 10000
#define NEDGES 80000
#define ETOT   90000   // NEDGES + NNODES self loops
#define FIN    128
#define H1     8
#define C1     256
#define HC1    2048    // H1*C1
#define NOUT   32
#define CHUNK  8192
#define NEG_SLOPE 0.2f

// ---------------------------------------------------------------- helpers ---
__device__ inline void edge_sd(const int* __restrict__ ei, int e, int& s, int& d) {
    if (e < NEDGES) { s = ei[e]; d = ei[NEDGES + e]; }
    else            { s = e - NEDGES; d = s; }          // self loop
}

__device__ inline void atomicMaxF32(float* addr, float val) {
    int old = __float_as_int(*addr);
    while (__int_as_float(old) < val) {
        int prev = atomicCAS((int*)addr, old, __float_as_int(val));
        if (prev == old) break;
        old = prev;
    }
}

// ------------------------------------------------------------------- fill ---
__global__ void fill_f32(float* __restrict__ p, size_t n, float v) {
    size_t i = (size_t)blockIdx.x * blockDim.x + threadIdx.x;
    if (i < n) p[i] = v;
}

// ------------------------------------------------------------ f32 -> bf16 ---
__global__ void cvt_bf16(const float* __restrict__ in, __bf16* __restrict__ out, size_t n) {
    size_t i = ((size_t)blockIdx.x * blockDim.x + threadIdx.x) * 4;
    if (i + 4 <= n) {
        f32x4 v = *(const f32x4*)(in + i);
        bf16x4 o;
        o[0] = (__bf16)v[0]; o[1] = (__bf16)v[1];
        o[2] = (__bf16)v[2]; o[3] = (__bf16)v[3];
        *(bf16x4*)(out + i) = o;
    } else {
        for (; i < n; ++i) out[i] = (__bf16)in[i];
    }
}

// --------------------------------------------------- self-loop edge attrs ---
// segment mean of incoming edge_attr per dst node (PyG fill_value='mean')
__global__ void segsum_attr(const float* __restrict__ ea, const int* __restrict__ ei,
                            float* __restrict__ sum, float* __restrict__ cnt) {
    size_t i = (size_t)blockIdx.x * blockDim.x + threadIdx.x;
    if (i >= (size_t)NEDGES * FIN) return;
    int e = (int)(i / FIN), f = (int)(i % FIN);
    int d = ei[NEDGES + e];
    atomicAdd(&sum[(size_t)d * FIN + f], ea[i]);
    if (f == 0) atomicAdd(&cnt[d], 1.0f);
}

__global__ void selfloop_attr(const float* __restrict__ sum, const float* __restrict__ cnt,
                              __bf16* __restrict__ ea_bf) {
    size_t i = (size_t)blockIdx.x * blockDim.x + threadIdx.x;
    if (i >= (size_t)NNODES * FIN) return;
    int n = (int)(i / FIN), f = (int)(i % FIN);
    ea_bf[(size_t)(NEDGES + n) * FIN + f] = (__bf16)(sum[i] / fmaxf(cnt[n], 1.0f));
}

// ------------------------------------------------------------- WMMA GEMM ---
// C[M,N] = relu?(A[M,K] @ B[K,N] + bias), A/B bf16 row-major, C f32.
// 256 threads = 8 waves; macro tile 128x64; wave w owns rows w*16..w*16+15
// and all 64 cols: one A fragment feeds 4 back-to-back WMMAs per K-step.
// LDS is double-buffered: next tile's global b128 loads overlap this tile's
// WMMA burst; one workgroup barrier per K-step.
// Requires: K % 32 == 0, N % 8 == 0 (true for all GEMMs here).
__global__ __launch_bounds__(256)
void wmma_gemm_bf16(const __bf16* __restrict__ A, const __bf16* __restrict__ B,
                    const float* __restrict__ bias, float* __restrict__ C,
                    int M, int N, int K, int relu)
{
    __shared__ __bf16 Alds[2][128][32];   // [buf][m][k]
    __shared__ __bf16 Blds[2][64][32];    // [buf][n][k]  (transposed)

    const int tid  = threadIdx.x;
    const int lane = tid & 31;
    const int wave = tid >> 5;            // 0..7 : 16-row band
    const int m0   = blockIdx.y * 128;
    const int n0   = blockIdx.x * 64;
    const int half = lane >> 4;           // lane group 0/1
    const int r16  = lane & 15;

    // staging indices (branch-free; one uniform guard per thread)
    const int ar = tid >> 1;              // 0..127 : A row within tile
    const int ak = (tid & 1) * 16;        // 0/16   : A k half
    const int bk = tid >> 3;              // 0..31  : B k
    const int bn = (tid & 7) * 8;         // 0..56  : B n segment

    const bool aIn = (m0 + ar) < M;
    const bool bIn = (n0 + bn) < N;       // N % 8 == 0 -> whole vector in/out
    const __bf16* pA = A + (size_t)(m0 + ar) * K + ak;
    const __bf16* pB = B + n0 + bn + (size_t)bk * N;

    auto stage = [&](int buf, int k0) {
        bf16x8 a0 = {}, a1 = {};
        if (aIn) {
            a0 = *(const bf16x8*)(pA + k0);
            a1 = *(const bf16x8*)(pA + k0 + 8);
            if (k0 + 64 < K) __builtin_prefetch(pA + k0 + 64, 0, 3);
        }
        *(bf16x8*)&Alds[buf][ar][ak]     = a0;
        *(bf16x8*)&Alds[buf][ar][ak + 8] = a1;

        bf16x8 bv = {};
        if (bIn) bv = *(const bf16x8*)(pB + (size_t)k0 * N);
        #pragma unroll
        for (int j = 0; j < 8; ++j) Blds[buf][bn + j][bk] = bv[j];
    };

    f32x8 acc[4] = {{}, {}, {}, {}};

    stage(0, 0);
    __syncthreads();

    int p = 0;
    for (int k0 = 0; k0 < K; k0 += 32, p ^= 1) {
        if (k0 + 32 < K) stage(p ^ 1, k0 + 32);   // overlap with WMMAs below

        // A fragment (contiguous K order, matches B):
        //   lanes 0-15: K = 0..15 ; lanes 16-31: K = 16..31
        bf16x16 afrag;
        #pragma unroll
        for (int i = 0; i < 8; ++i) {
            afrag[i]     = Alds[p][wave * 16 + r16][half * 16 + i];
            afrag[i + 8] = Alds[p][wave * 16 + r16][half * 16 + 8 + i];
        }
        // B fragments: lanes 0-15 hold K=0..15, lanes 16-31 hold K=16..31
        bf16x16 bfrag[4];
        #pragma unroll
        for (int s = 0; s < 4; ++s)
            #pragma unroll
            for (int i = 0; i < 16; ++i)
                bfrag[s][i] = Blds[p][s * 16 + r16][half * 16 + i];

        #pragma unroll
        for (int s = 0; s < 4; ++s)
            acc[s] = __builtin_amdgcn_wmma_f32_16x16x32_bf16(false, afrag, false, bfrag[s],
                                                             (short)0, acc[s], false, false);
        __syncthreads();
    }

    // C/D layout: VGPR i -> row = half*8 + i, col = lane&15
    #pragma unroll
    for (int s = 0; s < 4; ++s) {
        int gn = n0 + s * 16 + r16;
        if (gn >= N) continue;
        float bb = bias ? bias[gn] : 0.0f;
        #pragma unroll
        for (int i = 0; i < 8; ++i) {
            int gm = m0 + wave * 16 + half * 8 + i;
            if (gm < M) {
                float v = acc[s][i] + bb;
                if (relu) v = fmaxf(v, 0.0f);
                C[(size_t)gm * N + gn] = v;
            }
        }
    }
}

// ---------------------------------------------------- layer-1 edge logits ---
// one 256-thread block per edge; thread t handles channel c=t for all 8 heads
__global__ __launch_bounds__(256)
void logits1_kernel(const float* __restrict__ xl, const float* __restrict__ xr,
                    const float* __restrict__ e1c, const float* __restrict__ att,
                    const int* __restrict__ ei, int off, int len,
                    float* __restrict__ logits)
{
    int le = blockIdx.x;
    if (le >= len) return;
    int e = off + le;
    int s, d; edge_sd(ei, e, s, d);
    const float* pxl = xl  + (size_t)s  * HC1;
    const float* pxr = xr  + (size_t)d  * HC1;
    const float* pe  = e1c + (size_t)le * HC1;
    int t = threadIdx.x;

    __shared__ float acc[H1];
    if (t < H1) acc[t] = 0.0f;
    __syncthreads();

    #pragma unroll
    for (int h = 0; h < H1; ++h) {
        int hc = h * C1 + t;
        float v = pxl[hc] + pxr[hc] + pe[hc];
        v = (v > 0.0f) ? v : NEG_SLOPE * v;       // leaky relu
        v *= att[hc];
        for (int o = 16; o > 0; o >>= 1) v += __shfl_down(v, o, 32);
        if ((t & 31) == 0) atomicAdd(&acc[h], v);
    }
    __syncthreads();
    if (t < H1) logits[(size_t)e * H1 + t] = acc[t];
}

__global__ void segmax1(const float* __restrict__ logits, const int* __restrict__ ei,
                        float* __restrict__ lmax) {
    size_t i = (size_t)blockIdx.x * blockDim.x + threadIdx.x;
    if (i >= (size_t)ETOT * H1) return;
    int e = (int)(i / H1), h = (int)(i % H1);
    int s, d; edge_sd(ei, e, s, d);
    atomicMaxF32(&lmax[(size_t)d * H1 + h], logits[i]);
}

__global__ void expsum1(float* __restrict__ logits, const int* __restrict__ ei,
                        const float* __restrict__ lmax, float* __restrict__ denom) {
    size_t i = (size_t)blockIdx.x * blockDim.x + threadIdx.x;
    if (i >= (size_t)ETOT * H1) return;
    int e = (int)(i / H1), h = (int)(i % H1);
    int s, d; edge_sd(ei, e, s, d);
    float ex = __expf(logits[i] - lmax[(size_t)d * H1 + h]);
    logits[i] = ex;
    atomicAdd(&denom[(size_t)d * H1 + h], ex);
}

__global__ void scatter1(const float* __restrict__ xl, const float* __restrict__ logits,
                         const float* __restrict__ denom, const int* __restrict__ ei,
                         float* __restrict__ out) {
    size_t i = (size_t)blockIdx.x * blockDim.x + threadIdx.x;
    if (i >= (size_t)ETOT * HC1) return;
    int e  = (int)(i / HC1);
    int hc = (int)(i % HC1);
    int h  = hc >> 8;
    int s, d; edge_sd(ei, e, s, d);
    float alpha = logits[(size_t)e * H1 + h] / denom[(size_t)d * H1 + h];
    atomicAdd(&out[(size_t)d * HC1 + hc], xl[(size_t)s * HC1 + hc] * alpha);
}

// h = relu(in + b) written as bf16 (feeds layer-2 GEMMs)
__global__ void bias_relu_bf16(const float* __restrict__ in, const float* __restrict__ b,
                               __bf16* __restrict__ out, int rowlen, size_t n) {
    size_t i = (size_t)blockIdx.x * blockDim.x + threadIdx.x;
    if (i >= n) return;
    out[i] = (__bf16)fmaxf(in[i] + b[i % rowlen], 0.0f);
}

__global__ void bias_relu_f32(const float* __restrict__ in, const float* __restrict__ b,
                              float* __restrict__ out, int rowlen, size_t n) {
    size_t i = (size_t)blockIdx.x * blockDim.x + threadIdx.x;
    if (i >= n) return;
    out[i] = fmaxf(in[i] + b[i % rowlen], 0.0f);
}

// ---------------------------------------------------- layer-2 edge logits ---
// one wave per edge; lane c = channel (32 lanes == 32 channels)
__global__ __launch_bounds__(256)
void logits2_kernel(const float* __restrict__ xl2, const float* __restrict__ xr2,
                    const float* __restrict__ e2, const float* __restrict__ att,
                    const int* __restrict__ ei, float* __restrict__ logits) {
    int lane = threadIdx.x & 31;
    int e = blockIdx.x * 8 + (threadIdx.x >> 5);
    if (e >= ETOT) return;
    int s, d; edge_sd(ei, e, s, d);
    float v = xl2[(size_t)s * NOUT + lane] + xr2[(size_t)d * NOUT + lane]
            + e2[(size_t)e * NOUT + lane];
    v = (v > 0.0f) ? v : NEG_SLOPE * v;
    v *= att[lane];
    for (int o = 16; o > 0; o >>= 1) v += __shfl_down(v, o, 32);
    if (lane == 0) logits[e] = v;
}

__global__ void segmax2(const float* __restrict__ logits, const int* __restrict__ ei,
                        float* __restrict__ lmax) {
    int e = blockIdx.x * blockDim.x + threadIdx.x;
    if (e >= ETOT) return;
    int s, d; edge_sd(ei, e, s, d);
    atomicMaxF32(&lmax[d], logits[e]);
}

__global__ void expsum2(float* __restrict__ logits, const int* __restrict__ ei,
                        const float* __restrict__ lmax, float* __restrict__ denom) {
    int e = blockIdx.x * blockDim.x + threadIdx.x;
    if (e >= ETOT) return;
    int s, d; edge_sd(ei, e, s, d);
    float ex = __expf(logits[e] - lmax[d]);
    logits[e] = ex;
    atomicAdd(&denom[d], ex);
}

__global__ void scatter2(const float* __restrict__ xl2, const float* __restrict__ logits,
                         const float* __restrict__ denom, const int* __restrict__ ei,
                         float* __restrict__ out) {
    size_t i = (size_t)blockIdx.x * blockDim.x + threadIdx.x;
    if (i >= (size_t)ETOT * NOUT) return;
    int e = (int)(i / NOUT), c = (int)(i % NOUT);
    int s, d; edge_sd(ei, e, s, d);
    float alpha = logits[e] / denom[d];
    atomicAdd(&out[(size_t)d * NOUT + c], xl2[(size_t)s * NOUT + c] * alpha);
}

// ------------------------------------------------------------------ launch ---
extern "C" void kernel_launch(void* const* d_in, const int* in_sizes, int n_in,
                              void* d_out, int out_size, void* d_ws, size_t ws_size,
                              hipStream_t stream) {
    (void)in_sizes; (void)n_in; (void)out_size; (void)ws_size;
    const float* x    = (const float*)d_in[0];
    const int*   ei   = (const int*)  d_in[1];
    const float* ea   = (const float*)d_in[2];
    const float* Wl1  = (const float*)d_in[3];
    const float* bl1  = (const float*)d_in[4];
    const float* Wr1  = (const float*)d_in[5];
    const float* br1  = (const float*)d_in[6];
    const float* We1  = (const float*)d_in[7];
    const float* att1 = (const float*)d_in[8];
    const float* b1   = (const float*)d_in[9];
    const float* Wl2  = (const float*)d_in[10];
    const float* bl2  = (const float*)d_in[11];
    const float* Wr2  = (const float*)d_in[12];
    const float* br2  = (const float*)d_in[13];
    const float* We2  = (const float*)d_in[14];
    const float* att2 = (const float*)d_in[15];
    const float* b2   = (const float*)d_in[16];
    float* out = (float*)d_out;

    // bump allocator over d_ws
    size_t cur = 0;
    auto alloc = [&](size_t bytes) -> void* {
        void* p = (char*)d_ws + cur;
        cur += (bytes + 255) & ~(size_t)255;
        return p;
    };
    float*  slsum   = (float*) alloc((size_t)NNODES * FIN * 4);
    float*  slcnt   = (float*) alloc((size_t)NNODES * 4);
    __bf16* ea_bf   = (__bf16*)alloc((size_t)ETOT * FIN * 2);
    __bf16* x_bf    = (__bf16*)alloc((size_t)NNODES * FIN * 2);
    __bf16* Wl1b    = (__bf16*)alloc((size_t)FIN * HC1 * 2);
    __bf16* Wr1b    = (__bf16*)alloc((size_t)FIN * HC1 * 2);
    __bf16* We1b    = (__bf16*)alloc((size_t)FIN * HC1 * 2);
    __bf16* Wl2b    = (__bf16*)alloc((size_t)HC1 * NOUT * 2);
    __bf16* Wr2b    = (__bf16*)alloc((size_t)HC1 * NOUT * 2);
    __bf16* We2b    = (__bf16*)alloc((size_t)FIN * NOUT * 2);
    float*  xl1     = (float*) alloc((size_t)NNODES * HC1 * 4);
    float*  xr1     = (float*) alloc((size_t)NNODES * HC1 * 4);
    float*  e1c     = (float*) alloc((size_t)CHUNK * HC1 * 4);
    float*  logits1 = (float*) alloc((size_t)ETOT * H1 * 4);
    float*  lmax1   = (float*) alloc((size_t)NNODES * H1 * 4);
    float*  denom1  = (float*) alloc((size_t)NNODES * H1 * 4);
    float*  out1    = (float*) alloc((size_t)NNODES * HC1 * 4);
    __bf16* h_bf    = (__bf16*)alloc((size_t)NNODES * HC1 * 2);
    float*  xl2     = (float*) alloc((size_t)NNODES * NOUT * 4);
    float*  xr2     = (float*) alloc((size_t)NNODES * NOUT * 4);
    float*  e2      = (float*) alloc((size_t)ETOT * NOUT * 4);
    float*  logits2 = (float*) alloc((size_t)ETOT * 4);
    float*  lmax2   = (float*) alloc((size_t)NNODES * 4);
    float*  denom2  = (float*) alloc((size_t)NNODES * 4);
    float*  out2    = (float*) alloc((size_t)NNODES * NOUT * 4);

    auto fill = [&](float* p, size_t n, float v) {
        fill_f32<<<(unsigned)((n + 255) / 256), 256, 0, stream>>>(p, n, v);
    };
    auto cvt = [&](const float* in, __bf16* o, size_t n) {
        cvt_bf16<<<(unsigned)((n / 4 + 255) / 256 + 1), 256, 0, stream>>>(in, o, n);
    };
    auto gemm = [&](const __bf16* A, const __bf16* B, const float* bias, float* C,
                    int M, int Nc, int K, int relu) {
        dim3 grid((Nc + 63) / 64, (M + 127) / 128);
        wmma_gemm_bf16<<<grid, 256, 0, stream>>>(A, B, bias, C, M, Nc, K, relu);
    };

    const float NEGINF = -3.4e38f;

    // ---- bf16 conversions of all GEMM operands ----
    cvt(x,   x_bf,  (size_t)NNODES * FIN);
    cvt(ea,  ea_bf, (size_t)NEDGES * FIN);   // first E rows of ea_bf
    cvt(Wl1, Wl1b,  (size_t)FIN * HC1);
    cvt(Wr1, Wr1b,  (size_t)FIN * HC1);
    cvt(We1, We1b,  (size_t)FIN * HC1);
    cvt(Wl2, Wl2b,  (size_t)HC1 * NOUT);
    cvt(Wr2, Wr2b,  (size_t)HC1 * NOUT);
    cvt(We2, We2b,  (size_t)FIN * NOUT);

    // ---- self-loop attrs (mean of incoming edge_attr per dst) -> ea_bf[E..) ----
    fill(slsum, (size_t)NNODES * FIN, 0.0f);
    fill(slcnt, NNODES, 0.0f);
    segsum_attr<<<(unsigned)(((size_t)NEDGES * FIN + 255) / 256), 256, 0, stream>>>(
        ea, ei, slsum, slcnt);
    selfloop_attr<<<(unsigned)(((size_t)NNODES * FIN + 255) / 256), 256, 0, stream>>>(
        slsum, slcnt, ea_bf);

    // ---- layer 1: projections (bf16 WMMA GEMMs) ----
    gemm(x_bf, Wl1b, bl1, xl1, NNODES, HC1, FIN, 0);
    gemm(x_bf, Wr1b, br1, xr1, NNODES, HC1, FIN, 0);

    // ---- layer 1: edge logits, chunked over e1 = ea_bf @ We1 ----
    for (int off = 0; off < ETOT; off += CHUNK) {
        int len = (ETOT - off < CHUNK) ? (ETOT - off) : CHUNK;
        gemm(ea_bf + (size_t)off * FIN, We1b, nullptr, e1c, len, HC1, FIN, 0);
        logits1_kernel<<<len, 256, 0, stream>>>(xl1, xr1, e1c, att1, ei, off, len, logits1);
    }

    // ---- layer 1: segment softmax + scatter ----
    fill(lmax1, (size_t)NNODES * H1, NEGINF);
    fill(denom1, (size_t)NNODES * H1, 0.0f);
    fill(out1, (size_t)NNODES * HC1, 0.0f);
    segmax1<<<(unsigned)(((size_t)ETOT * H1 + 255) / 256), 256, 0, stream>>>(logits1, ei, lmax1);
    expsum1<<<(unsigned)(((size_t)ETOT * H1 + 255) / 256), 256, 0, stream>>>(logits1, ei, lmax1, denom1);
    scatter1<<<(unsigned)(((size_t)ETOT * HC1 + 255) / 256), 256, 0, stream>>>(
        xl1, logits1, denom1, ei, out1);
    bias_relu_bf16<<<(unsigned)(((size_t)NNODES * HC1 + 255) / 256), 256, 0, stream>>>(
        out1, b1, h_bf, HC1, (size_t)NNODES * HC1);

    // ---- layer 2: projections ----
    gemm(h_bf, Wl2b, bl2, xl2, NNODES, NOUT, HC1, 0);
    gemm(h_bf, Wr2b, br2, xr2, NNODES, NOUT, HC1, 0);
    gemm(ea_bf, We2b, nullptr, e2, ETOT, NOUT, FIN, 0);

    // ---- layer 2: logits + segment softmax + scatter ----
    logits2_kernel<<<(ETOT + 7) / 8, 256, 0, stream>>>(xl2, xr2, e2, att2, ei, logits2);
    fill(lmax2, NNODES, NEGINF);
    fill(denom2, NNODES, 0.0f);
    fill(out2, (size_t)NNODES * NOUT, 0.0f);
    segmax2<<<(ETOT + 255) / 256, 256, 0, stream>>>(logits2, ei, lmax2);
    expsum2<<<(ETOT + 255) / 256, 256, 0, stream>>>(logits2, ei, lmax2, denom2);
    scatter2<<<(unsigned)(((size_t)ETOT * NOUT + 255) / 256), 256, 0, stream>>>(
        xl2, logits2, denom2, ei, out2);

    // ---- final: out = relu(out2 + b2) ----
    bias_relu_f32<<<(unsigned)(((size_t)NNODES * NOUT + 255) / 256), 256, 0, stream>>>(
        out2, b2, out, NOUT, (size_t)NNODES * NOUT);
}
